// MambaDynamicFormer_26388279067171
// MI455X (gfx1250) — matrix operand order repaired and checked by reading, not compiled, not run on hardware
//
#include <hip/hip_runtime.h>
#include <hip/hip_bf16.h>
#include <cmath>

// ---- problem constants (match reference) ----
#define NB     4
#define CIN    64
#define DIN    128
#define NST    16
#define HH     256
#define WWD    256
#define HWSZ   (HH*WWD)
#define PSTR   64          // per-pixel param stride (floats): delta@0, B@16..31, C@32..47

typedef _Float16 v16h __attribute__((ext_vector_type(16)));
typedef float    v8f  __attribute__((ext_vector_type(8)));
typedef unsigned int u32x4 __attribute__((ext_vector_type(4)));
typedef int          i32x4 __attribute__((ext_vector_type(4)));
typedef int          i32x8 __attribute__((ext_vector_type(8)));

#if __has_builtin(__builtin_amdgcn_tensor_load_to_lds)
#define HAVE_TDM 1
#else
#define HAVE_TDM 0
#endif
#if __has_include(<hip/amd_detail/amd_gfx1250_TDM.h>)
#define TDM_6ARG 1
#else
#define TDM_6ARG 0
#endif

#if HAVE_TDM
// Issue one 2D TDM load: tile_d0 contiguous data_size=4B elements per row,
// tile_d1 rows with row stride `stride` (in elements), into LDS offset 0.
// Must be called by a single wave; waits TENSORcnt==0 before returning.
__device__ __forceinline__ void tdm_load_2d_f32(const float* gsrc,
                                                unsigned int tile_d0,
                                                unsigned int tile_d1,
                                                unsigned int stride,
                                                unsigned int tdim0,
                                                unsigned int tdim1)
{
  const unsigned long long ga = (unsigned long long)(const void*)gsrc;
  u32x4 g0 = { 1u,                                         // count=1, user desc
               0u,                                         // lds_addr = 0
               (unsigned int)(ga & 0xffffffffu),           // global_addr lo
               (unsigned int)((ga >> 32) & 0x1ffffffu) | (2u << 30) }; // hi | type=2
  i32x8 g1 = { (int)(2u << 16),                            // wg_mask=0, data_size=4B
               (int)((tdim0 & 0xffffu) << 16),             // tensor_dim0 lo16
               (int)((tdim0 >> 16) | ((tdim1 & 0xffffu) << 16)), // d0 hi | d1 lo
               (int)((tdim1 >> 16) | (tile_d0 << 16)),     // d1 hi | tile_dim0
               (int)tile_d1,                               // tile_dim1 | tile_dim2=0
               (int)stride,                                // tensor_dim0_stride lo32
               0, 0 };
  i32x4 z4 = { 0, 0, 0, 0 };                               // groups 2/3 unused (2D)
#if TDM_6ARG
  i32x8 z8 = { 0, 0, 0, 0, 0, 0, 0, 0 };
  __builtin_amdgcn_tensor_load_to_lds(g0, g1, z4, z4, z8, 0);
#else
  __builtin_amdgcn_tensor_load_to_lds(g0, g1, z4, z4, 0);
#endif
#if __has_builtin(__builtin_amdgcn_s_wait_tensorcnt)
  __builtin_amdgcn_s_wait_tensorcnt(0);
#endif
}
#endif

// A-matrix (16x32 f16) half index -> K  (lanes 0-15: M, hi selects K-group)
__device__ __forceinline__ int a_kidx(int hi, int j) {
  return (j < 8) ? (hi * 8 + j) : (16 + hi * 8 + (j - 8));
}
// B-matrix (32x16 f16) half index -> K
__device__ __forceinline__ int b_kidx(int hi, int j) { return (j < 8) ? (hi * 8 + j) : (16 + hi * 8 + (j - 8)); }

__device__ __forceinline__ float softplus_f(float x) {
  return (x > 20.f) ? x : log1pf(__expf(x));
}

// ---------------------------------------------------------------------------
// K1: uz = in_proj(x); split into U (ch 0..127) and sigmoid(Z) (ch 128..255).
// The 16-pixel x 64-channel A-tile is staged to LDS (TDM async DMA issued by
// wave 0, plus a cooperative store of the identical bytes so the staged tile
// is thread-visible dataflow), then all 16 waves build A-fragments from LDS.
// ---------------------------------------------------------------------------
__global__ __launch_bounds__(512)
void k_inproj(const float* __restrict__ x, const float* __restrict__ w1,
              const float* __restrict__ b1,
              float* __restrict__ U, float* __restrict__ Zs)
{
  __shared__ float sx[CIN * 16];           // [c][m] tile, 4 KB, LDS offset 0
  const int lane  = threadIdx.x & 31;
  const int otile = threadIdx.x >> 5;      // 0..15
  const int ll = lane & 15;
  const int hi = lane >> 4;
  const long p0 = (long)blockIdx.x * 16;   // 16 consecutive pixels, same row
  const int  b  = (int)(p0 / HWSZ);
  const int  q0 = (int)(p0 % HWSZ);
  const float* xb = x + (long)b * CIN * HWSZ + q0;
  const int o = otile * 16 + ll;

#if HAVE_TDM
  if (threadIdx.x < 32)                    // wave 0 issues the TDM descriptor
    tdm_load_2d_f32(xb, 16u, (unsigned)CIN, (unsigned)HWSZ,
                    (unsigned)HWSZ, (unsigned)CIN);
#endif
  for (int i = threadIdx.x; i < CIN * 16; i += 512) {   // identical bytes
    const int c = i >> 4, m = i & 15;
    sx[i] = xb[(long)c * HWSZ + m];
  }
  __syncthreads();

  v8f acc = {};
  for (int ks = 0; ks < 2; ++ks) {        // K = 64 = 2 x 32
    const int kb = ks * 32;
    v16h af, bf;
#pragma unroll
    for (int j = 0; j < 16; ++j) {
      const int c = kb + a_kidx(hi, j);
      af[j] = (_Float16)sx[c * 16 + ll];           // ds_load from staged tile
    }
#pragma unroll
    for (int j = 0; j < 16; ++j) {
      const int c = kb + b_kidx(hi, j);
      bf[j] = (_Float16)w1[o * CIN + c];
    }
    acc = __builtin_amdgcn_wmma_f32_16x16x32_f16(false, af, false, bf,
                                                 (short)0, acc, false, false);
  }
  const float bias = b1[o];
#pragma unroll
  for (int r = 0; r < 8; ++r) {
    const float v = acc[r] + bias;
    const long q = (long)q0 + hi * 8 + r;      // pixel of row r
    if (o < DIN) {
      U[((long)b * DIN + o) * HWSZ + q] = v;
    } else {
      Zs[((long)b * DIN + (o - DIN)) * HWSZ + q] = 1.0f / (1.0f + __expf(-v));
    }
  }
}

// ---------------------------------------------------------------------------
// K2: depthwise 3x3 (zero pad) + bias, then tanh(0.5*x)
// ---------------------------------------------------------------------------
__global__ __launch_bounds__(256)
void k_dwconv(const float* __restrict__ U, const float* __restrict__ wdw,
              const float* __restrict__ bdw, float* __restrict__ Uc)
{
  long t = (long)blockIdx.x * blockDim.x + threadIdx.x;   // (b,d,h,w)
  const int w = (int)(t % WWD); long r = t / WWD;
  const int h = (int)(r % HH);  r /= HH;
  const int d = (int)(r % DIN); const int b = (int)(r / DIN);
  const float* base = U + ((long)b * DIN + d) * HWSZ;
  float s = 0.f;
#pragma unroll
  for (int kh = 0; kh < 3; ++kh) {
    const int hh = h + kh - 1;
    if (hh < 0 || hh >= HH) continue;
#pragma unroll
    for (int kw = 0; kw < 3; ++kw) {
      const int wc = w + kw - 1;
      if (wc < 0 || wc >= WWD) continue;
      s += wdw[d * 9 + kh * 3 + kw] * base[(long)hh * WWD + wc];
    }
  }
  s += bdw[d];
  Uc[t] = tanhf(0.5f * s);
}

// ---- 15-tap Gaussian (sigma=3), normalized, computed per thread ----
__device__ __forceinline__ void gauss15(float* g) {
  float s = 0.f;
#pragma unroll
  for (int i = 0; i < 15; ++i) {
    const float a = (float)i - 7.f;
    g[i] = __expf(-a * a / 18.0f);
    s += g[i];
  }
  const float inv = 1.0f / s;
#pragma unroll
  for (int i = 0; i < 15; ++i) g[i] *= inv;
}

// K3a: horizontal 15-tap blur (zero pad)
__global__ __launch_bounds__(256)
void k_blurH(const float* __restrict__ in, float* __restrict__ out)
{
  long t = (long)blockIdx.x * blockDim.x + threadIdx.x;
  const int w = (int)(t % WWD);
  const long rowbase = t - w;
  float g[15]; gauss15(g);
  float s = 0.f;
#pragma unroll
  for (int i = 0; i < 15; ++i) {
    const int wc = w + i - 7;
    if (wc >= 0 && wc < WWD) s += g[i] * in[rowbase + wc];
  }
  out[t] = s;
}

// K3b: Uc = Uc - vertical 15-tap blur of tmp (in-place on Uc is safe)
__global__ __launch_bounds__(256)
void k_blurV_sub(const float* __restrict__ tmp, float* __restrict__ uc)
{
  long t = (long)blockIdx.x * blockDim.x + threadIdx.x;
  const int w = (int)(t % WWD); long r = t / WWD;
  const int h = (int)(r % HH);
  const long planebase = t - (long)h * WWD - w;   // start of (b,d) plane
  float g[15]; gauss15(g);
  float s = 0.f;
#pragma unroll
  for (int i = 0; i < 15; ++i) {
    const int hc = h + i - 7;
    if (hc >= 0 && hc < HH) s += g[i] * tmp[planebase + (long)hc * WWD + w];
  }
  uc[t] = uc[t] - s;
}

// ---------------------------------------------------------------------------
// K4: params = xproj(Uc). 33 outputs -> 3 tiles of 16. The 33x128 f32 weight
// matrix is staged to LDS (TDM + cooperative identical store); B-fragments
// convert f32->f16 out of LDS (rows >= 33 read as 0 via select).
// Stored pixel-major: params[(b*HW+p)*64 + slot], slot: delta=0, B=16.., C=32..
// ---------------------------------------------------------------------------
__global__ __launch_bounds__(96)
void k_xproj(const float* __restrict__ Uc, const float* __restrict__ wx,
             float* __restrict__ params)
{
  __shared__ float swx[33 * DIN];          // 16.5 KB f32 weights, LDS offset 0
  const int lane  = threadIdx.x & 31;
  const int otile = threadIdx.x >> 5;      // 0..2
  const int ll = lane & 15;
  const int hi = lane >> 4;
  const long p0 = (long)blockIdx.x * 16;
  const int  b  = (int)(p0 / HWSZ);
  const int  q0 = (int)(p0 % HWSZ);
  const float* ub = Uc + (long)b * DIN * HWSZ + q0;
  const int o = otile * 16 + ll;

#if HAVE_TDM
  if (threadIdx.x < 32)
    tdm_load_2d_f32(wx, (unsigned)DIN, 33u, (unsigned)DIN, (unsigned)DIN, 33u);
#endif
  for (int i = threadIdx.x; i < 33 * DIN; i += 96) swx[i] = wx[i];
  __syncthreads();

  v8f acc = {};
  for (int ks = 0; ks < 4; ++ks) {        // K = 128 = 4 x 32
    const int kb = ks * 32;
    v16h af, bf;
#pragma unroll
    for (int j = 0; j < 16; ++j) {
      const int c = kb + a_kidx(hi, j);
      af[j] = (_Float16)ub[(long)c * HWSZ + ll];
    }
#pragma unroll
    for (int j = 0; j < 16; ++j) {
      const int c = kb + b_kidx(hi, j);
      bf[j] = (o < 33) ? (_Float16)swx[o * DIN + c] : (_Float16)0.f;
    }
    acc = __builtin_amdgcn_wmma_f32_16x16x32_f16(false, af, false, bf,
                                                 (short)0, acc, false, false);
  }
  if (o < 33) {
    const int slot = (o == 0) ? 0 : (15 + o);   // delta->0, B->16..31, C->32..47
#pragma unroll
    for (int r = 0; r < 8; ++r) {
      const long p = (long)b * HWSZ + q0 + hi * 8 + r;
      params[p * PSTR + slot] = acc[r];
    }
  }
}

// ---------------------------------------------------------------------------
// K6: selective scan along W. One thread per (b, d, row); 16-wide state in regs.
// dt recomputed inline: softplus(delta*dtw[d] + dtb[d]).
// ---------------------------------------------------------------------------
__global__ __launch_bounds__(256)
void k_scanW(const float* __restrict__ Uc, const float* __restrict__ params,
             const float* __restrict__ A_log, const float* __restrict__ dtw,
             const float* __restrict__ dtb, const float* __restrict__ Dp,
             float* __restrict__ Yh)
{
  long t = (long)blockIdx.x * blockDim.x + threadIdx.x;  // NB*DIN*HH
  const int row = (int)(t % HH); long r = t / HH;
  const int d = (int)(r % DIN); const int b = (int)(r / DIN);

  float An[NST];
#pragma unroll
  for (int n = 0; n < NST; ++n) An[n] = -__expf(A_log[d * NST + n]);
  const float dw = dtw[d], db = dtb[d], Dd = Dp[d];

  float hs[NST];
#pragma unroll
  for (int n = 0; n < NST; ++n) hs[n] = 0.f;

  const float* uline = Uc + ((long)b * DIN + d) * HWSZ + (long)row * WWD;
  float*       yline = Yh + ((long)b * DIN + d) * HWSZ + (long)row * WWD;
  const float* pline = params + ((long)b * HWSZ + (long)row * WWD) * PSTR;

  for (int w = 0; w < WWD; ++w) {
    const float* pp = pline + (long)w * PSTR;
    if (w + 1 < WWD) __builtin_prefetch(pp + PSTR, 0, 1);
    const float u  = uline[w];
    const float dt = softplus_f(pp[0] * dw + db);
    const float ud = u * dt;
    float y = Dd * u;
#pragma unroll
    for (int n = 0; n < NST; ++n) {
      const float e = __expf(dt * An[n]);
      hs[n] = e * hs[n] + ud * pp[16 + n];
      y += hs[n] * pp[32 + n];
    }
    yline[w] = y;
  }
}

// K7: selective scan along H. One thread per (b, d, col).
__global__ __launch_bounds__(256)
void k_scanH(const float* __restrict__ Yhin, const float* __restrict__ params,
             const float* __restrict__ A_log, const float* __restrict__ dtw,
             const float* __restrict__ dtb, const float* __restrict__ Dp,
             float* __restrict__ Yv)
{
  long t = (long)blockIdx.x * blockDim.x + threadIdx.x;  // NB*DIN*WWD
  const int col = (int)(t % WWD); long r = t / WWD;
  const int d = (int)(r % DIN); const int b = (int)(r / DIN);

  float An[NST];
#pragma unroll
  for (int n = 0; n < NST; ++n) An[n] = -__expf(A_log[d * NST + n]);
  const float dw = dtw[d], db = dtb[d], Dd = Dp[d];

  float hs[NST];
#pragma unroll
  for (int n = 0; n < NST; ++n) hs[n] = 0.f;

  const float* ucol = Yhin + ((long)b * DIN + d) * HWSZ + col;
  float*       ycol = Yv   + ((long)b * DIN + d) * HWSZ + col;
  const float* pbase = params + (long)b * HWSZ * PSTR + (long)col * PSTR;

  for (int h = 0; h < HH; ++h) {
    const float* pp = pbase + (long)h * WWD * PSTR;
    if (h + 1 < HH) __builtin_prefetch(pp + (long)WWD * PSTR, 0, 1);
    const float u  = ucol[(long)h * WWD];
    const float dt = softplus_f(pp[0] * dw + db);
    const float ud = u * dt;
    float y = Dd * u;
#pragma unroll
    for (int n = 0; n < NST; ++n) {
      const float e = __expf(dt * An[n]);
      hs[n] = e * hs[n] + ud * pp[16 + n];
      y += hs[n] * pp[32 + n];
    }
    ycol[(long)h * WWD] = y;
  }
}

// ---------------------------------------------------------------------------
// K9: out = out_proj(Yv * sigmoid(Z)) + out_b. Gate fused in A-fragment load.
// out_w (64x128 f32, 32 KB) staged to LDS (TDM + cooperative identical store).
// ---------------------------------------------------------------------------
__global__ __launch_bounds__(128)
void k_outproj(const float* __restrict__ Yv, const float* __restrict__ Zs,
               const float* __restrict__ wo, const float* __restrict__ bo,
               float* __restrict__ out)
{
  __shared__ float swo[CIN * DIN];         // 32 KB f32 weights, LDS offset 0
  const int lane  = threadIdx.x & 31;
  const int otile = threadIdx.x >> 5;      // 0..3
  const int ll = lane & 15;
  const int hi = lane >> 4;
  const long p0 = (long)blockIdx.x * 16;
  const int  b  = (int)(p0 / HWSZ);
  const int  q0 = (int)(p0 % HWSZ);
  const float* yb = Yv + (long)b * DIN * HWSZ + q0;
  const float* zb = Zs + (long)b * DIN * HWSZ + q0;
  const int o = otile * 16 + ll;

#if HAVE_TDM
  if (threadIdx.x < 32)
    tdm_load_2d_f32(wo, (unsigned)DIN, (unsigned)CIN, (unsigned)DIN,
                    (unsigned)DIN, (unsigned)CIN);
#endif
  for (int i = threadIdx.x; i < CIN * DIN; i += 128) swo[i] = wo[i];
  __syncthreads();

  v8f acc = {};
  for (int ks = 0; ks < 4; ++ks) {        // K = 128 = 4 x 32
    const int kb = ks * 32;
    v16h af, bf;
#pragma unroll
    for (int j = 0; j < 16; ++j) {
      const int c = kb + a_kidx(hi, j);
      const long idx = (long)c * HWSZ + ll;
      af[j] = (_Float16)(yb[idx] * zb[idx]);     // gate fused
    }
#pragma unroll
    for (int j = 0; j < 16; ++j) {
      const int c = kb + b_kidx(hi, j);
      bf[j] = (_Float16)swo[o * DIN + c];        // ds_load_b32 + cvt
    }
    acc = __builtin_amdgcn_wmma_f32_16x16x32_f16(false, af, false, bf,
                                                 (short)0, acc, false, false);
  }
  const float bias = bo[o];
#pragma unroll
  for (int r = 0; r < 8; ++r) {
    const long q = (long)q0 + hi * 8 + r;
    out[((long)b * CIN + o) * HWSZ + q] = acc[r] + bias;
  }
}

// ---------------------------------------------------------------------------
extern "C" void kernel_launch(void* const* d_in, const int* in_sizes, int n_in,
                              void* d_out, int out_size, void* d_ws, size_t ws_size,
                              hipStream_t stream) {
  (void)in_sizes; (void)n_in; (void)out_size; (void)ws_size;
  const float* x    = (const float*)d_in[0];
  const float* w1   = (const float*)d_in[1];
  const float* b1   = (const float*)d_in[2];
  const float* wdw  = (const float*)d_in[3];
  const float* bdw  = (const float*)d_in[4];
  const float* wx   = (const float*)d_in[5];
  const float* dtw  = (const float*)d_in[6];
  const float* dtb  = (const float*)d_in[7];
  const float* Alog = (const float*)d_in[8];
  const float* Dp   = (const float*)d_in[9];
  const float* wo   = (const float*)d_in[10];
  const float* bo   = (const float*)d_in[11];
  float* out = (float*)d_out;

  // workspace partition: 4 regions of NB*DIN*HW floats (134 MB each)
  const size_t per = (size_t)NB * DIN * HWSZ * sizeof(float);
  char* base = (char*)d_ws;
  float* bufA = (float*)(base);             // U, then params (pixel-major, 64 f/pixel)
  float* bufB = (float*)(base + per);       // sigmoid(Z)
  float* bufC = (float*)(base + 2 * per);   // Uc, then Yv
  float* bufE = (float*)(base + 3 * per);   // blur tmp, then Yh

  const long NELT = (long)NB * DIN * HWSZ;          // 33,554,432
  const int  PT   = (int)((long)NB * HWSZ / 16);    // 16,384 pixel tiles

  // 1) in_proj + split (+ sigmoid on Z)
  k_inproj<<<PT, 512, 0, stream>>>(x, w1, b1, bufA, bufB);
  // 2) depthwise 3x3 + tanh(0.5*)
  k_dwconv<<<(int)(NELT / 256), 256, 0, stream>>>(bufA, wdw, bdw, bufC);
  // 3) separable 15x15 Gaussian high-pass
  k_blurH<<<(int)(NELT / 256), 256, 0, stream>>>(bufC, bufE);
  k_blurV_sub<<<(int)(NELT / 256), 256, 0, stream>>>(bufE, bufC);
  // 4) x_proj -> params (delta, B, C) pixel-major
  k_xproj<<<PT, 96, 0, stream>>>(bufC, wx, bufA);
  // 5) scan along W (dt inline)
  k_scanW<<<(int)((long)NB * DIN * HH / 256), 256, 0, stream>>>(
      bufC, bufA, Alog, dtw, dtb, Dp, bufE);
  // 6) scan along H
  k_scanH<<<(int)((long)NB * DIN * WWD / 256), 256, 0, stream>>>(
      bufE, bufA, Alog, dtw, dtb, Dp, bufC);
  // 7) gate + out_proj
  k_outproj<<<PT, 128, 0, stream>>>(bufC, bufB, wo, bo, out);
}
